// RNN_11879879543742
// MI455X (gfx1250) — matrix operand order repaired and checked by reading, not compiled, not run on hardware
//
#include <hip/hip_runtime.h>
#include <hip/hip_bf16.h>

// Problem constants
#define T_STEPS 32
#define BATCH   32
#define NIN     64
#define HID     256
#define NOUT    64
#define EMAX    10240          // 320 * 32 error rows at the end
#define MAXBLK  80             // max row-blocks (128 rows each) per sample
#define CLAMP_MIN (-2.0f)
#define CLAMP_MAX ( 2.0f)

typedef __attribute__((ext_vector_type(16))) __bf16 bf16x16;
typedef __attribute__((ext_vector_type(8)))  float  f32x8;
typedef unsigned int uintx4 __attribute__((ext_vector_type(4)));
typedef int          intx4  __attribute__((ext_vector_type(4)));
typedef int          intx8  __attribute__((ext_vector_type(8)));

union BFrag { uint4 u[2]; bf16x16 v; };

__device__ __forceinline__ unsigned short f2bf(float f) {
    union { float f; unsigned int u; } c; c.f = f;
    unsigned int r = ((c.u >> 16) & 1u) + 0x7FFFu;   // round-to-nearest-even
    return (unsigned short)((c.u + r) >> 16);
}
__device__ __forceinline__ float bf2f(unsigned short h) {
    union { unsigned int u; float f; } c; c.u = ((unsigned int)h) << 16;
    return c.f;
}
__device__ __forceinline__ unsigned int pack2(float a, float b) {
    return (unsigned int)f2bf(a) | ((unsigned int)f2bf(b) << 16);
}

// ---- Tensor Data Mover: load a [rows x 32] bf16 tile (row stride 256 elems)
// ---- from global into contiguous LDS at byte offset lds_off.  ISA §8.3/8.4.
__device__ __forceinline__ void tdm_load_slice(const unsigned short* gsrc,
                                               unsigned int lds_off,
                                               unsigned int rows) {
    unsigned long long ga = (unsigned long long)(size_t)gsrc;
    uintx4 g0 = { 1u,                                   // count=1, user descriptor
                  lds_off,                              // LDS byte address
                  (unsigned int)(ga & 0xFFFFFFFFu),     // global_addr[31:0]
                  ((unsigned int)((ga >> 32) & 0x01FFFFFFu)) | 0x80000000u }; // [56:32] | type=2
    intx8 g1;
    g1[0] = 0x00010000;                 // workgroup_mask=0, data_size=1 (2 bytes)
    g1[1] = (int)(32u << 16);           // tensor_dim0 = 32   (lo16 @ bits[31:16])
    g1[2] = (int)(rows << 16);          // tensor_dim0 hi=0, tensor_dim1 = rows (lo16)
    g1[3] = (int)(32u << 16);           // tensor_dim1 hi=0, tile_dim0 = 32
    g1[4] = (int)rows;                  // tile_dim1 = rows, tile_dim2 = 0
    g1[5] = 256;                        // tensor_dim0_stride = 256 elements (lo32)
    g1[6] = 0;                          // stride hi16 / tensor_dim1_stride lo16
    g1[7] = 0;
    intx4 z4 = {0, 0, 0, 0};
#if __clang_major__ >= 23
    intx8 z8 = {0, 0, 0, 0, 0, 0, 0, 0};
    __builtin_amdgcn_tensor_load_to_lds(g0, g1, z4, z4, z8, 0);
#else
    __builtin_amdgcn_tensor_load_to_lds(g0, g1, z4, z4, 0);
#endif
}

// ---------- one-time prep: bf16-transposed weights ----------
__global__ void conv_whhT(const float* __restrict__ Whh, unsigned short* __restrict__ WhhT) {
    int n = blockIdx.x, k = threadIdx.x;               // 256 x 256
    WhhT[n * HID + k] = f2bf(Whh[k * HID + n]);
}
__global__ void conv_woT(const float* __restrict__ Wo, unsigned short* __restrict__ WoT) {
    int n = blockIdx.x, k = threadIdx.x;               // 64 x 256
    WoT[n * HID + k] = f2bf(Wo[k * NOUT + n]);
}
__global__ void init_headh(float* __restrict__ head_h) {
    int i = blockIdx.x * 256 + threadIdx.x;
    if (i < BATCH * HID) head_h[i] = 0.f;
}

// ---------- per-step scalar path (fp32): interval, head GEMM, err_x radius ----------
__global__ __launch_bounds__(256)
void head_kernel(const float* __restrict__ Xt, const float* __restrict__ eps,
                 const float* __restrict__ Wih, const float* __restrict__ Whh,
                 const float* __restrict__ bih, const float* __restrict__ bhh,
                 const float* __restrict__ head_h,
                 float* __restrict__ head, float* __restrict__ rad, float* __restrict__ dbuf) {
    __shared__ float sx[NIN], sd[NIN];
    int b = blockIdx.x, tid = threadIdx.x;
    if (tid < NIN) {
        float x = Xt[b * NIN + tid];
        float e = eps[b];
        float l = fmaxf(x - e, CLAMP_MIN);
        float u = fminf(x + e, CLAMP_MAX);
        sx[tid] = 0.5f * (l + u);
        float dd = 0.5f * (u - l);
        sd[tid] = dd;
        dbuf[b * NIN + tid] = dd;
    }
    __syncthreads();
    int h = tid;
    float acc = bih[h] + bhh[h];
    float rx  = 0.f;
    #pragma unroll 8
    for (int i = 0; i < NIN; ++i) {
        float w = Wih[i * HID + h];
        acc += sx[i] * w;
        rx  += sd[i] * fabsf(w);           // row-wise |diag(d) @ W_ih| radius
    }
    #pragma unroll 8
    for (int k = 0; k < HID; ++k) acc += head_h[b * HID + k] * Whh[k * HID + h];
    head[b * HID + h] = acc;
    rad[b * HID + h]  = rx;
}

// ---------- big WMMA GEMM: errNew[0:Eh) = errOld @ W_hh ----------
// 8 waves x 16 rows = 128 rows x all 256 cols.  TDM double-buffers the
// W_hhT k-slice in LDS; B-fragment ds_loads are pipelined one tile ahead.
__global__ __launch_bounds__(256)
void step_gemm(const unsigned short* __restrict__ errOld,
               unsigned short* __restrict__ errNew,
               const unsigned short* __restrict__ WhhT,   // [n][k] row-major
               float* __restrict__ radPartial,            // [b][MAXBLK][HID]
               int Eh) {
    __shared__ unsigned short sB[2][HID * 32];  // 2 x 16 KB TDM-filled k-slices
    __shared__ float sred[8][HID];              // 8 KB per-wave |.| column sums
    const int b    = blockIdx.y;
    const int blk  = blockIdx.x;
    const int tid  = threadIdx.x;
    const int wave = tid >> 5;
    const int lane = tid & 31;
    const int row0 = blk * 128 + wave * 16;
    const bool active = (row0 < Eh);            // Eh is always a multiple of 16

    const unsigned short* Abase = errOld + ((size_t)b * EMAX + row0) * HID;
    const unsigned int sboff[2] = { (unsigned int)(size_t)(void*)&sB[0][0],
                                    (unsigned int)(size_t)(void*)&sB[1][0] };

    f32x8 acc[16];
    #pragma unroll
    for (int ct = 0; ct < 16; ++ct) acc[ct] = (f32x8){0.f,0.f,0.f,0.f,0.f,0.f,0.f,0.f};

    const int m   = lane & 15;
    const int kbA = (lane >> 4) * 16;           // A: contiguous K per lane half
    const int kbB = (lane >> 4) * 16;           // B: contiguous K per lane half

    if (wave == 0) {                            // prologue: slice 0 via TDM
        tdm_load_slice(WhhT, sboff[0], HID);
        __builtin_amdgcn_s_wait_tensorcnt(0);
    }
    __syncthreads();

    for (int kt = 0; kt < 8; ++kt) {
        const int k0 = kt * 32;
        if (wave == 0 && kt < 7)                // overlap: fetch slice kt+1
            tdm_load_slice(WhhT + (kt + 1) * 32, sboff[(kt + 1) & 1], HID);
        const unsigned short* sbuf = sB[kt & 1];
        if (active) {
            BFrag a;
            a.u[0] = *(const uint4*)(Abase + m * HID + k0 + kbA);
            a.u[1] = *(const uint4*)(Abase + m * HID + k0 + kbA + 8);
            if (kt < 7)
                __builtin_prefetch(Abase + m * HID + k0 + 32, 0, 0);
            BFrag bc, bn;
            {
                const uint4* bp = (const uint4*)(sbuf + m * 32 + kbB);
                bc.u[0] = bp[0]; bc.u[1] = bp[1];
            }
            #pragma unroll
            for (int ct = 0; ct < 16; ++ct) {
                if (ct < 15) {                  // pipeline next B frag over this WMMA
                    const uint4* bp = (const uint4*)(sbuf + ((ct + 1) * 16 + m) * 32 + kbB);
                    bn.u[0] = bp[0]; bn.u[1] = bp[1];
                }
                acc[ct] = __builtin_amdgcn_wmma_f32_16x16x32_bf16(
                    false, a.v, false, bc.v, (short)0, acc[ct], false, false);
                bc = bn;
            }
        }
        if (wave == 0) __builtin_amdgcn_s_wait_tensorcnt(0);
        __syncthreads();
    }

    if (active) {
        #pragma unroll
        for (int ct = 0; ct < 16; ++ct) {
            float s = 0.f;
            #pragma unroll
            for (int i = 0; i < 8; ++i) {
                float v = acc[ct][i];
                s += fabsf(v);
                int r = ((lane >> 4) * 8) + i;          // C layout: M split across lane halves
                int n = ct * 16 + m;
                errNew[((size_t)b * EMAX + row0 + r) * HID + n] = f2bf(v);
            }
            s += __shfl_xor(s, 16, 32);                 // fold the two row halves
            if (lane < 16) sred[wave][ct * 16 + lane] = s;
        }
    } else {
        for (int j = lane; j < HID; j += 32) sred[wave][j] = 0.f;
    }
    __syncthreads();
    {   // deterministic per-block partial radius (no float atomics)
        float t = 0.f;
        #pragma unroll
        for (int w = 0; w < 8; ++w) t += sred[w][tid];
        radPartial[((size_t)b * MAXBLK + blk) * HID + tid] = t;
    }
}

// ---------- tanh zonotope transformer (fp32) ----------
__global__ void lam_kernel(const float* __restrict__ head, const float* __restrict__ rad,
                           const float* __restrict__ radPartial,
                           float* __restrict__ lam, float* __restrict__ delta,
                           float* __restrict__ head_h, int nblk) {
    int b = blockIdx.x, h = threadIdx.x;
    float r = rad[b * HID + h];
    for (int j = 0; j < nblk; ++j) r += radPartial[((size_t)b * MAXBLK + j) * HID + h];
    float hd = head[b * HID + h];
    float l = hd - r, u = hd + r;
    float tl = tanhf(l), tu = tanhf(u);
    float la = fminf(1.f - tl * tl, 1.f - tu * tu);
    float mu = 0.5f * (tu + tl - la * (u + l));
    float de = 0.5f * (tu - tl - la * (u - l));
    lam[b * HID + h]    = la;
    delta[b * HID + h]  = de;
    head_h[b * HID + h] = la * hd + mu;
}

// ---------- scale old rows by lam, append err_x rows and diag(delta) rows ----------
// vectorized: one uint4 (8 bf16) per thread, 8 rows per 256-thread block.
__global__ __launch_bounds__(256)
void scale_append(unsigned short* __restrict__ err, const float* __restrict__ lam,
                  const float* __restrict__ delta, const float* __restrict__ dbuf,
                  const float* __restrict__ Wih, int Eh) {
    const int b  = blockIdx.y;
    const int r  = blockIdx.x * 8 + (threadIdx.x >> 5);
    const int c0 = (threadIdx.x & 31) * 8;
    const int hbase = b * HID + c0;
    uint4* p = (uint4*)(err + ((size_t)b * EMAX + r) * HID + c0);
    float vals[8];
    if (r < Eh) {                                        // lam * (err_h @ W_hh)
        uint4 v = *p;
        unsigned int u[4] = {v.x, v.y, v.z, v.w};
        #pragma unroll
        for (int j = 0; j < 4; ++j) {
            vals[2*j]   = bf2f((unsigned short)(u[j] & 0xFFFFu)) * lam[hbase + 2*j];
            vals[2*j+1] = bf2f((unsigned short)(u[j] >> 16))     * lam[hbase + 2*j + 1];
        }
    } else if (r < Eh + NIN) {                           // lam * (d_i * W_ih[i,:])
        int i = r - Eh;
        float dv = dbuf[b * NIN + i];
        #pragma unroll
        for (int j = 0; j < 8; ++j)
            vals[j] = lam[hbase + j] * dv * Wih[i * HID + c0 + j];
    } else {                                             // diag(delta)
        int jr = r - Eh - NIN;
        float de = delta[b * HID + jr];
        #pragma unroll
        for (int j = 0; j < 8; ++j)
            vals[j] = (c0 + j == jr) ? de : 0.f;
    }
    uint4 o;
    o.x = pack2(vals[0], vals[1]);
    o.y = pack2(vals[2], vals[3]);
    o.z = pack2(vals[4], vals[5]);
    o.w = pack2(vals[6], vals[7]);
    *p = o;
}

// ---------- final WMMA GEMM: row-|.| sums of errFinal @ W_o ----------
__global__ __launch_bounds__(256)
void out_gemm(const unsigned short* __restrict__ errF,
              const unsigned short* __restrict__ WoT,    // [n=64][k=256]
              float* __restrict__ radPartialOut) {       // [b][MAXBLK][NOUT]
    __shared__ unsigned short sB[2][NOUT * 32];  // 2 x 4 KB TDM-filled slices
    __shared__ float sred[8][NOUT];
    const int b    = blockIdx.y;
    const int blk  = blockIdx.x;
    const int tid  = threadIdx.x;
    const int wave = tid >> 5;
    const int lane = tid & 31;
    const int row0 = blk * 128 + wave * 16;      // EMAX/128 == 80 exact, all active

    const unsigned short* Abase = errF + ((size_t)b * EMAX + row0) * HID;
    const unsigned int sboff[2] = { (unsigned int)(size_t)(void*)&sB[0][0],
                                    (unsigned int)(size_t)(void*)&sB[1][0] };

    f32x8 acc[4];
    #pragma unroll
    for (int ct = 0; ct < 4; ++ct) acc[ct] = (f32x8){0.f,0.f,0.f,0.f,0.f,0.f,0.f,0.f};

    const int m   = lane & 15;
    const int kbA = (lane >> 4) * 16;
    const int kbB = (lane >> 4) * 16;

    if (wave == 0) {
        tdm_load_slice(WoT, sboff[0], NOUT);
        __builtin_amdgcn_s_wait_tensorcnt(0);
    }
    __syncthreads();

    for (int kt = 0; kt < 8; ++kt) {
        const int k0 = kt * 32;
        if (wave == 0 && kt < 7)
            tdm_load_slice(WoT + (kt + 1) * 32, sboff[(kt + 1) & 1], NOUT);
        const unsigned short* sbuf = sB[kt & 1];
        BFrag a;
        a.u[0] = *(const uint4*)(Abase + m * HID + k0 + kbA);
        a.u[1] = *(const uint4*)(Abase + m * HID + k0 + kbA + 8);
        if (kt < 7)
            __builtin_prefetch(Abase + m * HID + k0 + 32, 0, 0);
        BFrag bc, bn;
        {
            const uint4* bp = (const uint4*)(sbuf + m * 32 + kbB);
            bc.u[0] = bp[0]; bc.u[1] = bp[1];
        }
        #pragma unroll
        for (int ct = 0; ct < 4; ++ct) {
            if (ct < 3) {
                const uint4* bp = (const uint4*)(sbuf + ((ct + 1) * 16 + m) * 32 + kbB);
                bn.u[0] = bp[0]; bn.u[1] = bp[1];
            }
            acc[ct] = __builtin_amdgcn_wmma_f32_16x16x32_bf16(
                false, a.v, false, bc.v, (short)0, acc[ct], false, false);
            bc = bn;
        }
        if (wave == 0) __builtin_amdgcn_s_wait_tensorcnt(0);
        __syncthreads();
    }
    #pragma unroll
    for (int ct = 0; ct < 4; ++ct) {
        float s = 0.f;
        #pragma unroll
        for (int i = 0; i < 8; ++i) s += fabsf(acc[ct][i]);
        s += __shfl_xor(s, 16, 32);
        if (lane < 16) sred[wave][ct * 16 + lane] = s;
    }
    __syncthreads();
    if (tid < NOUT) {
        float t = 0.f;
        #pragma unroll
        for (int w = 0; w < 8; ++w) t += sred[w][tid];
        radPartialOut[((size_t)b * MAXBLK + blk) * NOUT + tid] = t;
    }
}

// ---------- final bounds: head_out +/- rad_out ----------
__global__ void out_kernel(const float* __restrict__ head_h, const float* __restrict__ Wo,
                           const float* __restrict__ bo, const float* __restrict__ radPartialOut,
                           float* __restrict__ out, int nblk) {
    int b = blockIdx.x, n = threadIdx.x;   // 32 x 64
    float acc = bo[n];
    #pragma unroll 8
    for (int k = 0; k < HID; ++k) acc += head_h[b * HID + k] * Wo[k * NOUT + n];
    float r = 0.f;
    for (int j = 0; j < nblk; ++j) r += radPartialOut[((size_t)b * MAXBLK + j) * NOUT + n];
    out[b * NOUT + n]                 = acc - r;   // lower
    out[BATCH * NOUT + b * NOUT + n]  = acc + r;   // upper
}

extern "C" void kernel_launch(void* const* d_in, const int* in_sizes, int n_in,
                              void* d_out, int out_size, void* d_ws, size_t ws_size,
                              hipStream_t stream) {
    const float* X    = (const float*)d_in[0];   // [T,B,NIN]
    const float* eps  = (const float*)d_in[1];   // [B]
    const float* Wih  = (const float*)d_in[2];   // [NIN,HID]
    const float* Whh  = (const float*)d_in[3];   // [HID,HID]
    const float* bih  = (const float*)d_in[4];   // [HID]
    const float* bhh  = (const float*)d_in[5];   // [HID]
    const float* Wo   = (const float*)d_in[6];   // [HID,NOUT]
    const float* bo   = (const float*)d_in[7];   // [NOUT]
    float* out = (float*)d_out;

    size_t off = 0;
    auto alloc = [&](size_t bytes) -> void* {
        void* p = (char*)d_ws + off;
        off += (bytes + 255) & ~(size_t)255;
        return p;
    };
    unsigned short* errA = (unsigned short*)alloc((size_t)BATCH * EMAX * HID * 2);
    unsigned short* errB = (unsigned short*)alloc((size_t)BATCH * EMAX * HID * 2);
    unsigned short* WhhT = (unsigned short*)alloc((size_t)HID * HID * 2);
    unsigned short* WoT  = (unsigned short*)alloc((size_t)NOUT * HID * 2);
    float* head_h   = (float*)alloc(BATCH * HID * 4);
    float* head     = (float*)alloc(BATCH * HID * 4);
    float* rad      = (float*)alloc(BATCH * HID * 4);
    float* lam      = (float*)alloc(BATCH * HID * 4);
    float* delta    = (float*)alloc(BATCH * HID * 4);
    float* dbuf     = (float*)alloc(BATCH * NIN * 4);
    float* radPart  = (float*)alloc((size_t)BATCH * MAXBLK * HID * 4);
    float* radPartO = (float*)alloc((size_t)BATCH * MAXBLK * NOUT * 4);

    // prep
    conv_whhT<<<HID, HID, 0, stream>>>(Whh, WhhT);
    conv_woT<<<NOUT, HID, 0, stream>>>(Wo, WoT);
    init_headh<<<(BATCH * HID + 255) / 256, 256, 0, stream>>>(head_h);

    unsigned short* bufs[2] = {errA, errB};
    int cur = 0;
    for (int t = 0; t < T_STEPS; ++t) {
        const int Eh   = 320 * t;
        const int nblk = (Eh + 127) / 128;
        head_kernel<<<BATCH, 256, 0, stream>>>(X + (size_t)t * BATCH * NIN, eps, Wih, Whh,
                                               bih, bhh, head_h, head, rad, dbuf);
        if (Eh > 0)
            step_gemm<<<dim3(nblk, BATCH), 256, 0, stream>>>(bufs[cur], bufs[cur ^ 1],
                                                             WhhT, radPart, Eh);
        lam_kernel<<<BATCH, HID, 0, stream>>>(head, rad, radPart, lam, delta, head_h, nblk);
        scale_append<<<dim3((Eh + 320) / 8, BATCH), 256, 0, stream>>>(bufs[cur ^ 1], lam, delta,
                                                                      dbuf, Wih, Eh);
        cur ^= 1;
    }

    // output layer: concretized bounds
    out_gemm<<<dim3(EMAX / 128, BATCH), 256, 0, stream>>>(bufs[cur], WoT, radPartO);
    out_kernel<<<BATCH, NOUT, 0, stream>>>(head_h, Wo, bo, radPartO, out, EMAX / 128);
}
